// ARX_RNN_ES_11020886081839
// MI455X (gfx1250) — matrix operand order repaired and checked by reading, not compiled, not run on hardware
//
#include <hip/hip_runtime.h>
#include <hip/hip_bf16.h>

// ARX RNN + exponential smoothing for MI455X (gfx1250, wave32, WMMA).
//
// Batch-independent recurrence: each block owns 16 batch rows, runs all 255
// steps locally, ONE barrier per step.
//
// Key algebraic trick: for autoregressive steps,
//   preact = x0*Wx0 + pred_prev*Wx1 + b + h_prev Wh
//          = x0*Wx0 + (b + bd*Wx1) + h_prev (Wh + Wd (x) Wx1)
// so after warmup we fold the rank-1 update Wd(x)Wx1 into Wh (f16, in LDS,
// B-fragments re-loaded to registers) and bd*Wx1 into the bias. The recurrence
// then depends ONLY on h via WMMA; the h.Wd readout (for the smoothed output)
// is background work consumed one step later by wave 0 only.
//
// 4 waves/block; wave w owns columns [32w,32w+32) = two 16-col N-tiles with
// independent WMMA accumulator chains; Wh^T B-fragments resident in VGPRs.

#define B_TOTAL  4096
#define T_TOTAL  256
#define LAGS     32
#define H        128
#define F        2
#define MROWS    16                  // batch rows per block (one WMMA M tile)
#define NWAVES   4                   // waves per block; each owns 2 N-tiles
#define NTHREADS (NWAVES * 32)
#define T_OUT    (T_TOTAL - LAGS)    // 224 output steps
#define NSTEPS   (T_TOTAL - 1)       // 255 recurrence steps

typedef __attribute__((ext_vector_type(16))) _Float16 v16h;
typedef __attribute__((ext_vector_type(8)))  float    v8f;

union V16H {
    v16h  v;
    uint4 q[2];
};

#if __has_builtin(__builtin_amdgcn_tanhf)
#define DEV_TANH(x) __builtin_amdgcn_tanhf(x)
#elif __has_builtin(__builtin_amdgcn_tanh_f32)
#define DEV_TANH(x) __builtin_amdgcn_tanh_f32(x)
#else
#define DEV_TANH(x) tanhf(x)
#endif

__global__ __launch_bounds__(NTHREADS)
void arx_rnn_es_kernel(const float* __restrict__ gIn,   // [B,T,F]
                       const float* __restrict__ gWx,   // [F,H]
                       const float* __restrict__ gWh,   // [H,H]
                       const float* __restrict__ gB,    // [H]
                       const float* __restrict__ gWd,   // [H,1]
                       const float* __restrict__ gBd,   // [1]
                       const float* __restrict__ gAl,   // [1]
                       float*       __restrict__ gOut)  // [B,T_OUT]
{
    // ---- LDS (~59 KB static; 2 blocks co-resident per WGP)
    __shared__ alignas(16) _Float16 sWhT[H * H];             // Wh^T f16: [n][k]
    __shared__ float    sX0[MROWS * T_TOTAL];                // exog feature 0
    __shared__ float    sX1[MROWS * LAGS];                   // feature 1, warmup
    __shared__ alignas(16) _Float16 sH[2][MROWS * H];        // ping-pong h (f16)
    __shared__ alignas(16) float sPredPart[2][MROWS][NWAVES];// dbl-buffered h.Wd

    const int tid  = threadIdx.x;
    const int wave = tid >> 5;      // 0..3 : owns columns [32w, 32w+32)
    const int lane = tid & 31;
    const int ln   = lane & 15;     // n within tile (B/C/D) or row m (A)
    const int hl   = lane >> 4;     // half-wave select
    const int r0   = blockIdx.x * MROWS;

    // ---- stage exogenous inputs for our 16 rows
    for (int i = tid; i < MROWS * T_TOTAL; i += NTHREADS) {
        int m = i >> 8, t = i & (T_TOTAL - 1);
        sX0[i] = gIn[((size_t)(r0 + m) * T_TOTAL + t) * F + 0];
    }
    for (int i = tid; i < MROWS * LAGS; i += NTHREADS) {
        int m = i >> 5, t = i & (LAGS - 1);
        sX1[i] = gIn[((size_t)(r0 + m) * T_TOTAL + t) * F + 1];
    }
    // ---- stage Wh^T as f16 (contiguous ds_load_b128 B-fragment fetch)
    for (int i = tid; i < H * H; i += NTHREADS) {
        int n = i >> 7, k = i & (H - 1);
        sWhT[i] = (_Float16)gWh[k * H + n];
    }
    // ---- h0 = 0
    for (int i = tid; i < MROWS * H; i += NTHREADS) sH[0][i] = (_Float16)0.0f;
    __syncthreads();

    // ---- per-lane constants for this wave's two columns n0, n1
    const int   n0    = wave * 32 + ln;
    const int   n1    = n0 + 16;
    const float bn0   = gB[n0],          bn1  = gB[n1];
    const float wx00  = gWx[0 * H + n0], wx01 = gWx[0 * H + n1];
    const float wx10  = gWx[1 * H + n0], wx11 = gWx[1 * H + n1];
    const float wdn0  = gWd[n0],         wdn1 = gWd[n1];
    const float alpha = fminf(fmaxf(gAl[0], 0.0f), 1.0f);
    const float bd    = gBd[0];

    // ---- B fragments (Wh^T) resident in VGPRs.
    // B 32x16 f16, k order used here (A matches): lane = column n; halves 0..7 = K 32f+16*hl..+7,
    // halves 8..15 = K 32f+16*hl+8..+15 (contiguous k, not the ISA table order).
    V16H bf0[4], bf1[4];
#pragma unroll
    for (int f = 0; f < 4; ++f) {
        const _Float16* p0 = &sWhT[n0 * H + f * 32 + hl * 16];
        const _Float16* p1 = &sWhT[n1 * H + f * 32 + hl * 16];
        bf0[f].q[0] = *(const uint4*)(p0);
        bf0[f].q[1] = *(const uint4*)(p0 + 8);
        bf1[f].q[0] = *(const uint4*)(p1);
        bf1[f].q[1] = *(const uint4*)(p1 + 8);
    }

    float es  = 0.0f;                // ES state: wave 0 lanes 0..15, m = lane
    int   cur = 0;

    // ================= warmup: t = 0 .. LAGS-1 (x from staged inputs) =======
    for (int t = 0; t < LAGS; ++t) {
        v8f c0, c1;
#pragma unroll
        for (int v = 0; v < 8; ++v) {
            const int   m  = v + hl * 8;
            const float x0 = sX0[m * T_TOTAL + t];
            const float x1 = sX1[m * LAGS + t];
            c0[v] = fmaf(x1, wx10, fmaf(x0, wx00, bn0));
            c1[v] = fmaf(x1, wx11, fmaf(x0, wx01, bn1));
        }
        V16H af[4];                  // A 16x32 f16: row m = ln
#pragma unroll
        for (int f = 0; f < 4; ++f) {
            const _Float16* p = &sH[cur][ln * H + f * 32 + hl * 16];
            af[f].q[0] = *(const uint4*)(p);
            af[f].q[1] = *(const uint4*)(p + 8);
        }
#pragma unroll
        for (int f = 0; f < 4; ++f) {   // two independent accumulator chains
            c0 = __builtin_amdgcn_wmma_f32_16x16x32_f16(
                     false, af[f].v, false, bf0[f].v, (short)0, c0, false, false);
            c1 = __builtin_amdgcn_wmma_f32_16x16x32_f16(
                     false, af[f].v, false, bf1[f].v, (short)0, c1, false, false);
        }
        float h0r[8], h1r[8];
#pragma unroll
        for (int v = 0; v < 8; ++v) {
            const float h0 = DEV_TANH(c0[v]);
            const float h1 = DEV_TANH(c1[v]);
            const int   m  = v + hl * 8;
            sH[cur ^ 1][m * H + n0] = (_Float16)h0;
            sH[cur ^ 1][m * H + n1] = (_Float16)h1;
            h0r[v] = h0; h1r[v] = h1;
        }
        // readout partials only needed for the last warmup step (first pred)
        if (t == LAGS - 1) {
            float s[8];
#pragma unroll
            for (int v = 0; v < 8; ++v) {
                s[v] = fmaf(h0r[v], wdn0, h1r[v] * wdn1);
#pragma unroll
                for (int mask = 8; mask >= 1; mask >>= 1)
                    s[v] += __shfl_xor(s[v], mask, 32);
            }
            if (ln == 0) {
#pragma unroll
                for (int v = 0; v < 8; ++v)
                    sPredPart[t & 1][hl * 8 + v][wave] = s[v];
            }
        }
        cur ^= 1;
        __syncthreads();
    }

    // ====== fold pred feedback into the weights: Wh' = Wh + Wd (x) Wx1 ======
    // (B-fragments are register-resident, so sWhT can be rebuilt freely.)
    for (int i = tid; i < H * H; i += NTHREADS) {
        int n = i >> 7, k = i & (H - 1);
        sWhT[i] = (_Float16)(gWh[k * H + n] + gWd[k] * gWx[H + n]);
    }
    __syncthreads();
#pragma unroll
    for (int f = 0; f < 4; ++f) {
        const _Float16* p0 = &sWhT[n0 * H + f * 32 + hl * 16];
        const _Float16* p1 = &sWhT[n1 * H + f * 32 + hl * 16];
        bf0[f].q[0] = *(const uint4*)(p0);
        bf0[f].q[1] = *(const uint4*)(p0 + 8);
        bf1[f].q[0] = *(const uint4*)(p1);
        bf1[f].q[1] = *(const uint4*)(p1 + 8);
    }
    const float bnA0 = fmaf(bd, wx10, bn0);   // fold bd*Wx1 into bias
    const float bnA1 = fmaf(bd, wx11, bn1);

    // ================= autoregressive: t = LAGS .. NSTEPS-1 =================
    for (int t = LAGS; t < NSTEPS; ++t) {
        const int buf  = t & 1;      // partials produced this iteration
        const int pbuf = buf ^ 1;    // partials from iteration t-1

        // ---- wave 0: emit smoothed output for step t-1 (background work)
        if (wave == 0 && lane < MROWS) {
            const float4 q = *(const float4*)&sPredPart[pbuf][lane][0];
            const float  pr = bd + q.x + q.y + q.z + q.w;
            es = (t == LAGS) ? pr : fmaf(alpha, pr, (1.0f - alpha) * es);
            gOut[(size_t)(r0 + lane) * T_OUT + (t - LAGS)] = es;
        }

        // ---- C = b' + x0_t*Wx0  (pred feedback already folded into Wh')
        v8f c0, c1;
#pragma unroll
        for (int v = 0; v < 8; ++v) {
            const float x0 = sX0[(v + hl * 8) * T_TOTAL + t];
            c0[v] = fmaf(x0, wx00, bnA0);
            c1[v] = fmaf(x0, wx01, bnA1);
        }

        // ---- A fragments + 8 WMMAs (two independent accumulator chains)
        V16H af[4];
#pragma unroll
        for (int f = 0; f < 4; ++f) {
            const _Float16* p = &sH[cur][ln * H + f * 32 + hl * 16];
            af[f].q[0] = *(const uint4*)(p);
            af[f].q[1] = *(const uint4*)(p + 8);
        }
#pragma unroll
        for (int f = 0; f < 4; ++f) {
            c0 = __builtin_amdgcn_wmma_f32_16x16x32_f16(
                     false, af[f].v, false, bf0[f].v, (short)0, c0, false, false);
            c1 = __builtin_amdgcn_wmma_f32_16x16x32_f16(
                     false, af[f].v, false, bf1[f].v, (short)0, c1, false, false);
        }

        // ---- tanh, publish h_{t+1} (f16), background h.Wd readout
        float s[8];
#pragma unroll
        for (int v = 0; v < 8; ++v) {
            const float h0 = DEV_TANH(c0[v]);
            const float h1 = DEV_TANH(c1[v]);
            const int   m  = v + hl * 8;
            sH[cur ^ 1][m * H + n0] = (_Float16)h0;
            sH[cur ^ 1][m * H + n1] = (_Float16)h1;
            s[v] = fmaf(h0, wdn0, h1 * wdn1);
        }
#pragma unroll
        for (int v = 0; v < 8; ++v) {
#pragma unroll
            for (int mask = 8; mask >= 1; mask >>= 1)
                s[v] += __shfl_xor(s[v], mask, 32);
        }
        if (ln == 0) {
#pragma unroll
            for (int v = 0; v < 8; ++v)
                sPredPart[buf][hl * 8 + v][wave] = s[v];
        }

        cur ^= 1;
        __syncthreads();             // the ONLY barrier per step
    }

    // ---- epilogue: output for the final step (t = NSTEPS-1)
    if (wave == 0 && lane < MROWS) {
        const int buf = (NSTEPS - 1) & 1;
        const float4 q = *(const float4*)&sPredPart[buf][lane][0];
        const float  pr  = bd + q.x + q.y + q.z + q.w;
        const float  esf = fmaf(alpha, pr, (1.0f - alpha) * es);
        gOut[(size_t)(r0 + lane) * T_OUT + (T_OUT - 1)] = esf;
    }
}

extern "C" void kernel_launch(void* const* d_in, const int* in_sizes, int n_in,
                              void* d_out, int out_size, void* d_ws, size_t ws_size,
                              hipStream_t stream) {
    (void)in_sizes; (void)n_in; (void)out_size; (void)d_ws; (void)ws_size;
    const float* gIn = (const float*)d_in[0];
    const float* gWx = (const float*)d_in[1];
    const float* gWh = (const float*)d_in[2];
    const float* gB  = (const float*)d_in[3];
    const float* gWd = (const float*)d_in[4];
    const float* gBd = (const float*)d_in[5];
    const float* gAl = (const float*)d_in[6];
    // d_in[7] = lags (compile-time constant LAGS=32 here)
    float* gOut = (float*)d_out;

    dim3 grid(B_TOTAL / MROWS);   // 256 independent batch tiles
    dim3 block(NTHREADS);         // 4 wave32 waves
    arx_rnn_es_kernel<<<grid, block, 0, stream>>>(gIn, gWx, gWh, gB, gWd, gBd,
                                                  gAl, gOut);
}
